// ParticleGNN_83820581749136
// MI455X (gfx1250) — matrix operand order repaired and checked by reading, compiled, not run on hardware
//
#include <hip/hip_runtime.h>
#include <hip/hip_bf16.h>
#include <math.h>

// ---- problem constants (match reference) ----
#define N_NODES 20000
#define N_EDGES 640000
#define N_GRAPH 64
#define D_IN    8
#define H_DIM   256
#define N_HEADS 4
#define D_HEAD  64

typedef __attribute__((ext_vector_type(16))) __bf16 v16bf;
typedef __attribute__((ext_vector_type(8)))  __bf16 v8bf;
typedef __attribute__((ext_vector_type(8)))  float  v8f;

#define GF_ACCUM 1
#define GF_RELU  2
#define GF_GELU  4

__device__ __forceinline__ float gelu_exact(float v) {
  return 0.5f * v * (1.0f + erff(v * 0.70710678118654752440f));
}
__device__ __forceinline__ float leaky02(float v) { return v > 0.f ? v : 0.2f * v; }

__device__ __forceinline__ void atomic_max_f32(float* addr, float val) {
  int* ai = (int*)addr;
  int old = __float_as_int(*addr);
  while (__int_as_float(old) < val) {
    int assumed = old;
    old = atomicCAS(ai, assumed, __float_as_int(val));
    if (old == assumed) break;
  }
}

// ---------- fp32 -> bf16 conversion helpers (feed the WMMA path) ----------
__global__ void k_cvt_bf16(const float* __restrict__ in, __bf16* __restrict__ out, int n) {
  int t = blockIdx.x * blockDim.x + threadIdx.x;
  if (t >= n) return;
  out[t] = (__bf16)in[t];
}
// W[K][N] fp32  ->  Wt[N][K] bf16  (K-transposed so B fragments are contiguous)
__global__ void k_cvt_bf16_tr(const float* __restrict__ W, __bf16* __restrict__ Wt,
                              int Kd, int Nd) {
  int t = blockIdx.x * blockDim.x + threadIdx.x;
  if (t >= Kd * Nd) return;
  int k = t / Nd, n = t % Nd;
  Wt[(size_t)n * Kd + k] = (__bf16)W[t];
}

// ======================= bf16 WMMA GEMM (L2-streaming, no LDS) =============
// C[M x Nd] = act( (ACCUM ? C : 0) + A[M x Kd] @ Bt[Nd x Kd]^T + bias )
// A bf16 row-major, Bt bf16 K-transposed. Block = 256 threads (8 waves),
// block tile 16(M) x 128(N); each wave owns one 16x16 tile; K unrolled by 64
// -> 8 global_load_b128 + 2 v_wmma per iteration, zero barriers.
__global__ __launch_bounds__(256)
void gemm_bf16_wmma(const __bf16* __restrict__ A, const __bf16* __restrict__ Bt,
                    float* __restrict__ C, const float* __restrict__ bias,
                    int M, int Kd, int Nd, int flags)
{
  const int lane = threadIdx.x & 31;
  const int wv   = threadIdx.x >> 5;
  const int mt   = blockIdx.x * 16;
  const int m    = lane & 15;
  const int hi   = lane >> 4;                 // lane half selects K/M split
  const int ncol = blockIdx.y * 128 + wv * 16 + m;

  // A fragment rows: lane-lo K {0..7,16..23}; lane-hi K {8..15,24..31}
  const __bf16* arow = A  + (size_t)(mt + m) * Kd + hi * 8;
  // B fragment rows: lane-lo K {0..7,16..23}; lane-hi K {8..15,24..31}
  const __bf16* brow = Bt + (size_t)ncol * Kd + hi * 8;

  v8f acc = {0.f, 0.f, 0.f, 0.f, 0.f, 0.f, 0.f, 0.f};
  if (flags & GF_ACCUM) {
#pragma unroll
    for (int r = 0; r < 8; ++r)
      acc[r] = C[(size_t)(mt + r + hi * 8) * Nd + ncol];
  }

  for (int k0 = 0; k0 < Kd; k0 += 64) {
    v8bf a0 = *(const v8bf*)(arow + k0);
    v8bf a1 = *(const v8bf*)(arow + k0 + 16);
    v8bf b0 = *(const v8bf*)(brow + k0);
    v8bf b1 = *(const v8bf*)(brow + k0 + 16);
    v8bf a2 = *(const v8bf*)(arow + k0 + 32);
    v8bf a3 = *(const v8bf*)(arow + k0 + 48);
    v8bf b2 = *(const v8bf*)(brow + k0 + 32);
    v8bf b3 = *(const v8bf*)(brow + k0 + 48);
    v16bf av0 = __builtin_shufflevector(a0, a1, 0,1,2,3,4,5,6,7,8,9,10,11,12,13,14,15);
    v16bf bv0 = __builtin_shufflevector(b0, b1, 0,1,2,3,4,5,6,7,8,9,10,11,12,13,14,15);
    acc = __builtin_amdgcn_wmma_f32_16x16x32_bf16(false, av0, false, bv0,
                                                  (short)0, acc, false, false);
    v16bf av1 = __builtin_shufflevector(a2, a3, 0,1,2,3,4,5,6,7,8,9,10,11,12,13,14,15);
    v16bf bv1 = __builtin_shufflevector(b2, b3, 0,1,2,3,4,5,6,7,8,9,10,11,12,13,14,15);
    acc = __builtin_amdgcn_wmma_f32_16x16x32_bf16(false, av1, false, bv1,
                                                  (short)0, acc, false, false);
  }

#pragma unroll
  for (int r = 0; r < 8; ++r) {
    float v = acc[r];
    if (bias) v += bias[ncol];
    if (flags & GF_RELU) v = fmaxf(v, 0.f);
    if (flags & GF_GELU) v = gelu_exact(v);
    C[(size_t)(mt + r + hi * 8) * Nd + ncol] = v;
  }
}

// ======================= elementwise / graph kernels =====================
#define GID ((int)(blockIdx.x * blockDim.x + threadIdx.x))

__global__ void k_encoder(const float* __restrict__ x, const float* __restrict__ W,
                          const float* __restrict__ b, float* __restrict__ out) {
  int t = GID; if (t >= N_NODES * H_DIM) return;
  int n = t >> 8, d = t & 255;
  float s = b[d];
#pragma unroll
  for (int i = 0; i < D_IN; ++i) s += x[n * D_IN + i] * W[i * H_DIM + d];
  out[t] = gelu_exact(s);
}

__global__ void k_count(const int* __restrict__ batch, float* __restrict__ cnt) {
  int t = GID; if (t >= N_NODES) return;
  atomicAdd(&cnt[batch[t]], 1.0f);
}

__global__ void k_gat_att(const float* __restrict__ hg, const float* __restrict__ as_,
                          const float* __restrict__ ad_, float* __restrict__ als,
                          float* __restrict__ ald) {
  int t = GID; if (t >= N_NODES * N_HEADS) return;
  int n = t >> 2, h = t & 3;
  const float* hp = hg + (size_t)n * H_DIM + h * D_HEAD;
  const float* s  = as_ + h * D_HEAD;
  const float* dd = ad_ + h * D_HEAD;
  float ss = 0.f, sd = 0.f;
#pragma unroll 8
  for (int j = 0; j < D_HEAD; ++j) { float v = hp[j]; ss += v * s[j]; sd += v * dd[j]; }
  als[t] = ss; ald[t] = sd;
}

__global__ void k_gat_minit(const float* als, const float* ald, float* m) {
  int t = GID; if (t >= N_NODES * N_HEADS) return;
  m[t] = leaky02(als[t] + ald[t]);        // self loop seeds the per-dst max
}

__global__ void k_gat_emax(const int* __restrict__ src, const int* __restrict__ dst,
                           const float* als, const float* ald, float* m) {
  int t = GID; if (t >= N_EDGES * N_HEADS) return;
  int e = t >> 2, h = t & 3;
  int s = src[e], d = dst[e];
  atomic_max_f32(&m[d * 4 + h], leaky02(als[s * 4 + h] + ald[d * 4 + h]));
}

__global__ void k_gat_sinit(const float* als, const float* ald, const float* m,
                            float* ssum) {
  int t = GID; if (t >= N_NODES * N_HEADS) return;
  ssum[t] = expf(leaky02(als[t] + ald[t]) - m[t]);   // self-loop term
}

__global__ void k_gat_eexp(const int* __restrict__ src, const int* __restrict__ dst,
                           const float* als, const float* ald, const float* m,
                           float* __restrict__ pbuf, float* __restrict__ ssum) {
  int t = GID; if (t >= N_EDGES * N_HEADS) return;
  int e = t >> 2, h = t & 3;
  int s = src[e], d = dst[e];
  float p = expf(leaky02(als[s * 4 + h] + ald[d * 4 + h]) - m[d * 4 + h]);
  pbuf[t] = p;
  atomicAdd(&ssum[d * 4 + h], p);
}

__global__ void k_gat_oinit(const float* __restrict__ hg, const float* als,
                            const float* ald, const float* m, const float* ssum,
                            float* __restrict__ out) {
  int t = GID; if (t >= N_NODES * H_DIM) return;
  int n = t >> 8, d = t & 255, h = d >> 6;
  float a = expf(leaky02(als[n * 4 + h] + ald[n * 4 + h]) - m[n * 4 + h]) /
            (ssum[n * 4 + h] + 1e-16f);
  out[t] = hg[t] * a;                     // self-loop contribution
}

__global__ void k_gat_agg(const int* __restrict__ src, const int* __restrict__ dst,
                          const float* __restrict__ hg, const float* __restrict__ pbuf,
                          const float* __restrict__ ssum, float* __restrict__ out) {
  int e = blockIdx.x, d = threadIdx.x;
  int h = d >> 6;
  int s = src[e], dn = dst[e];
  float a = pbuf[e * 4 + h] / (ssum[dn * 4 + h] + 1e-16f);
  atomicAdd(&out[(size_t)dn * H_DIM + d], hg[(size_t)s * H_DIM + d] * a);
}

__global__ void k_gn_sum(const float* __restrict__ x, const float* in_bias,
                         const int* __restrict__ batch, float* __restrict__ gsum) {
  int t = GID; if (t >= N_NODES * H_DIM) return;
  int n = t >> 8, d = t & 255;
  atomicAdd(&gsum[batch[n] * H_DIM + d], x[t] + (in_bias ? in_bias[d] : 0.f));
}

__global__ void k_gn_center(const float* __restrict__ x, const float* in_bias,
                            const int* __restrict__ batch, const float* gsum,
                            const float* cnt, const float* a,
                            float* __restrict__ tmp, float* __restrict__ gvar) {
  int t = GID; if (t >= N_NODES * H_DIM) return;
  int n = t >> 8, d = t & 255;
  int g = batch[n];
  float c = fmaxf(cnt[g], 1.f);
  float v = x[t] + (in_bias ? in_bias[d] : 0.f) - a[d] * (gsum[g * H_DIM + d] / c);
  tmp[t] = v;
  atomicAdd(&gvar[g * H_DIM + d], v * v);
}

__global__ void k_gn_final(float* __restrict__ tmp, const int* __restrict__ batch,
                           const float* gvar, const float* cnt,
                           const float* w, const float* b) {
  int t = GID; if (t >= N_NODES * H_DIM) return;
  int n = t >> 8, d = t & 255;
  int g = batch[n];
  float c = fmaxf(cnt[g], 1.f);
  float v = tmp[t] * rsqrtf(gvar[g * H_DIM + d] / c + 1e-5f) * w[d] + b[d];
  tmp[t] = fmaxf(v, 0.f);                 // both GraphNorm uses feed a relu
}

__global__ void k_deg(const int* __restrict__ dst, float* __restrict__ deg) {
  int t = GID; if (t >= N_EDGES) return;
  atomicAdd(&deg[dst[t]], 1.f);
}
__global__ void k_dis(const float* deg, float* dis) {
  int t = GID; if (t >= N_NODES) return;
  float d = deg[t];
  dis[t] = d > 0.f ? rsqrtf(fmaxf(d, 1.f)) : 0.f;
}
__global__ void k_tag_scatter(const int* __restrict__ src, const int* __restrict__ dst,
                              const float* __restrict__ dis, const float* __restrict__ in,
                              float* __restrict__ out) {
  int e = blockIdx.x, d = threadIdx.x;
  int s = src[e], dn = dst[e];
  float w = dis[s] * dis[dn];
  atomicAdd(&out[(size_t)dn * H_DIM + d], w * in[(size_t)s * H_DIM + d]);
}
__global__ void k_gc_scatter(const int* __restrict__ src, const int* __restrict__ dst,
                             const float* __restrict__ in, float* __restrict__ out) {
  int e = blockIdx.x, d = threadIdx.x;
  int s = src[e], dn = dst[e];
  atomicAdd(&out[(size_t)dn * H_DIM + d], in[(size_t)s * H_DIM + d]);
}

__global__ void k_pool_init(float* xmax) {
  int t = GID; if (t >= N_GRAPH * H_DIM) return;
  xmax[t] = -INFINITY;
}
__global__ void k_pool_scan(const float* __restrict__ x, const int* __restrict__ batch,
                            float* __restrict__ xmax, float* __restrict__ psum) {
  int t = GID; if (t >= N_NODES * H_DIM) return;
  int n = t >> 8, d = t & 255;
  int g = batch[n];
  float v = x[t];
  atomic_max_f32(&xmax[g * H_DIM + d], v);
  atomicAdd(&psum[g * H_DIM + d], v);
}
__global__ void k_pool_final(const float* xmax, const float* psum, const float* cnt,
                             float* __restrict__ pooled) {
  int t = GID; if (t >= N_GRAPH * H_DIM) return;
  int g = t >> 8, d = t & 255;
  float mx = xmax[t];
  if (!isfinite(mx)) mx = 0.f;
  pooled[(size_t)g * 512 + d]       = mx;
  pooled[(size_t)g * 512 + 256 + d] = psum[t] / fmaxf(cnt[g], 1.f);
}

__global__ void k_cls_final(const float* __restrict__ z1, const float* __restrict__ W2,
                            const float* __restrict__ b2, float* __restrict__ out) {
  int g = GID; if (g >= N_GRAPH) return;
  float z0 = b2[0], za = b2[1];
  const float* zp = z1 + (size_t)g * H_DIM;
  for (int j = 0; j < H_DIM; ++j) {
    float v = zp[j];
    z0 += v * W2[j * 2 + 0];
    za += v * W2[j * 2 + 1];
  }
  float mx = fmaxf(z0, za);
  float l = logf(expf(z0 - mx) + expf(za - mx));
  out[g * 2 + 0] = z0 - mx - l;
  out[g * 2 + 1] = za - mx - l;
}

// ============================ launcher ============================
static inline int cdiv_h(long long a, int b) { return (int)((a + b - 1) / b); }

extern "C" void kernel_launch(void* const* d_in, const int* in_sizes, int n_in,
                              void* d_out, int out_size, void* d_ws, size_t ws_size,
                              hipStream_t stream) {
  (void)in_sizes; (void)n_in; (void)out_size; (void)ws_size;
  const float* x      = (const float*)d_in[0];
  const int*   ei     = (const int*)d_in[1];
  const int*   batch  = (const int*)d_in[2];
  const float* W_enc  = (const float*)d_in[3];
  const float* b_enc  = (const float*)d_in[4];
  const float* W_gat  = (const float*)d_in[5];
  const float* att_s  = (const float*)d_in[6];
  const float* att_d  = (const float*)d_in[7];
  const float* b_gat  = (const float*)d_in[8];
  const float* W_tag  = (const float*)d_in[9];
  const float* b_tag  = (const float*)d_in[10];
  const float* W_rel  = (const float*)d_in[11];
  const float* b_rel  = (const float*)d_in[12];
  const float* W_root = (const float*)d_in[13];
  const float* g1_w   = (const float*)d_in[14];
  const float* g1_b   = (const float*)d_in[15];
  const float* g1_a   = (const float*)d_in[16];
  const float* g2_w   = (const float*)d_in[17];
  const float* g2_b   = (const float*)d_in[18];
  const float* g2_a   = (const float*)d_in[19];
  const float* W_c1   = (const float*)d_in[20];
  const float* b_c1   = (const float*)d_in[21];
  const float* W_c2   = (const float*)d_in[22];
  const float* b_c2   = (const float*)d_in[23];
  float* out = (float*)d_out;

  const int* src = ei;
  const int* dst = ei + N_EDGES;

  // workspace carve-up (256B-aligned chunks)
  float* wsp = (float*)d_ws;
  size_t off = 0;
  auto take  = [&](size_t n)  { float* p = wsp + off; off += (n + 63) & ~(size_t)63; return p; };
  auto takeb = [&](size_t nb) { return (__bf16*)take((nb + 1) / 2); };  // nb bf16 elems
  const size_t NH = (size_t)N_NODES * H_DIM;
  const int HH = H_DIM * H_DIM;
  float*  BUF0 = take(NH);
  float*  BUF1 = take(NH);
  float*  BUF2 = take(NH);
  __bf16* ABF  = takeb(NH);                 // bf16 copy of current GEMM A operand
  __bf16* WgatT  = takeb(HH);
  __bf16* Wtag0T = takeb(HH);
  __bf16* Wtag1T = takeb(HH);
  __bf16* Wtag2T = takeb(HH);
  __bf16* Wtag3T = takeb(HH);
  __bf16* WrelT  = takeb(HH);
  __bf16* WrootT = takeb(HH);
  __bf16* Wc1T   = takeb(2 * HH);
  __bf16* PBF    = takeb((size_t)N_GRAPH * 512);
  float* als  = take((size_t)N_NODES * N_HEADS);
  float* ald  = take((size_t)N_NODES * N_HEADS);
  float* mmax = take((size_t)N_NODES * N_HEADS);
  float* ssum = take((size_t)N_NODES * N_HEADS);
  float* pbuf = take((size_t)N_EDGES * N_HEADS);
  float* deg  = take(N_NODES);
  float* dis  = take(N_NODES);
  float* cnt  = take(N_GRAPH);
  float* gsum = take((size_t)N_GRAPH * H_DIM);
  float* gvar = take((size_t)N_GRAPH * H_DIM);
  float* xmax = take((size_t)N_GRAPH * H_DIM);
  float* psum = take((size_t)N_GRAPH * H_DIM);
  float* pooled = take((size_t)N_GRAPH * 512);
  float* z1   = take((size_t)N_GRAPH * H_DIM);

  const int TPB = 256;
  const dim3 gblk(256);
  const dim3 ggrid(N_NODES / 16, H_DIM / 128);      // 16(M) x 128(N) block tiles

  // ---- one-time per launch: K-transposed bf16 weights ----
  k_cvt_bf16_tr<<<cdiv_h(HH, TPB), TPB, 0, stream>>>(W_gat,          WgatT,  H_DIM, H_DIM);
  k_cvt_bf16_tr<<<cdiv_h(HH, TPB), TPB, 0, stream>>>(W_tag + 0 * HH, Wtag0T, H_DIM, H_DIM);
  k_cvt_bf16_tr<<<cdiv_h(HH, TPB), TPB, 0, stream>>>(W_tag + 1 * HH, Wtag1T, H_DIM, H_DIM);
  k_cvt_bf16_tr<<<cdiv_h(HH, TPB), TPB, 0, stream>>>(W_tag + 2 * HH, Wtag2T, H_DIM, H_DIM);
  k_cvt_bf16_tr<<<cdiv_h(HH, TPB), TPB, 0, stream>>>(W_tag + 3 * HH, Wtag3T, H_DIM, H_DIM);
  k_cvt_bf16_tr<<<cdiv_h(HH, TPB), TPB, 0, stream>>>(W_rel,          WrelT,  H_DIM, H_DIM);
  k_cvt_bf16_tr<<<cdiv_h(HH, TPB), TPB, 0, stream>>>(W_root,         WrootT, H_DIM, H_DIM);
  k_cvt_bf16_tr<<<cdiv_h(2 * HH, TPB), TPB, 0, stream>>>(W_c1,       Wc1T, 2 * H_DIM, H_DIM);

  // graph sizes
  hipMemsetAsync(cnt, 0, N_GRAPH * sizeof(float), stream);
  k_count<<<cdiv_h(N_NODES, TPB), TPB, 0, stream>>>(batch, cnt);

  // encoder: BUF0 = gelu(x @ W_enc + b_enc)
  k_encoder<<<cdiv_h((long long)NH, TPB), TPB, 0, stream>>>(x, W_enc, b_enc, BUF0);

  // GAT: BUF1 = BUF0 @ W_gat  (WMMA)
  k_cvt_bf16<<<cdiv_h((long long)NH, TPB), TPB, 0, stream>>>(BUF0, ABF, (int)NH);
  gemm_bf16_wmma<<<ggrid, gblk, 0, stream>>>(ABF, WgatT, BUF1, nullptr,
                                             N_NODES, H_DIM, H_DIM, 0);
  k_gat_att<<<cdiv_h((long long)N_NODES * N_HEADS, TPB), TPB, 0, stream>>>(BUF1, att_s, att_d, als, ald);
  k_gat_minit<<<cdiv_h((long long)N_NODES * N_HEADS, TPB), TPB, 0, stream>>>(als, ald, mmax);
  k_gat_emax<<<cdiv_h((long long)N_EDGES * N_HEADS, TPB), TPB, 0, stream>>>(src, dst, als, ald, mmax);
  k_gat_sinit<<<cdiv_h((long long)N_NODES * N_HEADS, TPB), TPB, 0, stream>>>(als, ald, mmax, ssum);
  k_gat_eexp<<<cdiv_h((long long)N_EDGES * N_HEADS, TPB), TPB, 0, stream>>>(src, dst, als, ald, mmax, pbuf, ssum);
  k_gat_oinit<<<cdiv_h((long long)NH, TPB), TPB, 0, stream>>>(BUF1, als, ald, mmax, ssum, BUF2);
  k_gat_agg<<<N_EDGES, H_DIM, 0, stream>>>(src, dst, BUF1, pbuf, ssum, BUF2);

  // GraphNorm1 (input BUF2 + b_gat) + relu -> BUF0
  hipMemsetAsync(gsum, 0, (size_t)N_GRAPH * H_DIM * sizeof(float), stream);
  hipMemsetAsync(gvar, 0, (size_t)N_GRAPH * H_DIM * sizeof(float), stream);
  k_gn_sum<<<cdiv_h((long long)NH, TPB), TPB, 0, stream>>>(BUF2, b_gat, batch, gsum);
  k_gn_center<<<cdiv_h((long long)NH, TPB), TPB, 0, stream>>>(BUF2, b_gat, batch, gsum, cnt, g1_a, BUF0, gvar);
  k_gn_final<<<cdiv_h((long long)NH, TPB), TPB, 0, stream>>>(BUF0, batch, gvar, cnt, g1_w, g1_b);

  // TAGConv -> BUF2   (4 WMMA GEMMs, 3 normalized scatters)
  hipMemsetAsync(deg, 0, N_NODES * sizeof(float), stream);
  k_deg<<<cdiv_h(N_EDGES, TPB), TPB, 0, stream>>>(dst, deg);
  k_dis<<<cdiv_h(N_NODES, TPB), TPB, 0, stream>>>(deg, dis);
  k_cvt_bf16<<<cdiv_h((long long)NH, TPB), TPB, 0, stream>>>(BUF0, ABF, (int)NH);
  gemm_bf16_wmma<<<ggrid, gblk, 0, stream>>>(ABF, Wtag0T, BUF2, b_tag,
                                             N_NODES, H_DIM, H_DIM, 0);
  hipMemsetAsync(BUF1, 0, NH * sizeof(float), stream);
  k_tag_scatter<<<N_EDGES, H_DIM, 0, stream>>>(src, dst, dis, BUF0, BUF1);
  k_cvt_bf16<<<cdiv_h((long long)NH, TPB), TPB, 0, stream>>>(BUF1, ABF, (int)NH);
  gemm_bf16_wmma<<<ggrid, gblk, 0, stream>>>(ABF, Wtag1T, BUF2, nullptr,
                                             N_NODES, H_DIM, H_DIM, GF_ACCUM);
  hipMemsetAsync(BUF0, 0, NH * sizeof(float), stream);
  k_tag_scatter<<<N_EDGES, H_DIM, 0, stream>>>(src, dst, dis, BUF1, BUF0);
  k_cvt_bf16<<<cdiv_h((long long)NH, TPB), TPB, 0, stream>>>(BUF0, ABF, (int)NH);
  gemm_bf16_wmma<<<ggrid, gblk, 0, stream>>>(ABF, Wtag2T, BUF2, nullptr,
                                             N_NODES, H_DIM, H_DIM, GF_ACCUM);
  hipMemsetAsync(BUF1, 0, NH * sizeof(float), stream);
  k_tag_scatter<<<N_EDGES, H_DIM, 0, stream>>>(src, dst, dis, BUF0, BUF1);
  k_cvt_bf16<<<cdiv_h((long long)NH, TPB), TPB, 0, stream>>>(BUF1, ABF, (int)NH);
  gemm_bf16_wmma<<<ggrid, gblk, 0, stream>>>(ABF, Wtag3T, BUF2, nullptr,
                                             N_NODES, H_DIM, H_DIM, GF_ACCUM);

  // GraphNorm2 (input BUF2) + relu -> BUF0
  hipMemsetAsync(gsum, 0, (size_t)N_GRAPH * H_DIM * sizeof(float), stream);
  hipMemsetAsync(gvar, 0, (size_t)N_GRAPH * H_DIM * sizeof(float), stream);
  k_gn_sum<<<cdiv_h((long long)NH, TPB), TPB, 0, stream>>>(BUF2, nullptr, batch, gsum);
  k_gn_center<<<cdiv_h((long long)NH, TPB), TPB, 0, stream>>>(BUF2, nullptr, batch, gsum, cnt, g2_a, BUF0, gvar);
  k_gn_final<<<cdiv_h((long long)NH, TPB), TPB, 0, stream>>>(BUF0, batch, gvar, cnt, g2_w, g2_b);

  // GraphConv -> BUF2 = relu(agg @ W_rel + b_rel + BUF0 @ W_root)
  hipMemsetAsync(BUF1, 0, NH * sizeof(float), stream);
  k_gc_scatter<<<N_EDGES, H_DIM, 0, stream>>>(src, dst, BUF0, BUF1);
  k_cvt_bf16<<<cdiv_h((long long)NH, TPB), TPB, 0, stream>>>(BUF1, ABF, (int)NH);
  gemm_bf16_wmma<<<ggrid, gblk, 0, stream>>>(ABF, WrelT, BUF2, b_rel,
                                             N_NODES, H_DIM, H_DIM, 0);
  k_cvt_bf16<<<cdiv_h((long long)NH, TPB), TPB, 0, stream>>>(BUF0, ABF, (int)NH);
  gemm_bf16_wmma<<<ggrid, gblk, 0, stream>>>(ABF, WrootT, BUF2, nullptr,
                                             N_NODES, H_DIM, H_DIM, GF_ACCUM | GF_RELU);

  // pooling -> pooled [64 x 512] = [max | mean]
  k_pool_init<<<cdiv_h((long long)N_GRAPH * H_DIM, TPB), TPB, 0, stream>>>(xmax);
  hipMemsetAsync(psum, 0, (size_t)N_GRAPH * H_DIM * sizeof(float), stream);
  k_pool_scan<<<cdiv_h((long long)NH, TPB), TPB, 0, stream>>>(BUF2, batch, xmax, psum);
  k_pool_final<<<cdiv_h((long long)N_GRAPH * H_DIM, TPB), TPB, 0, stream>>>(xmax, psum, cnt, pooled);

  // classifier: z1 = gelu(pooled @ W_c1 + b_c1)  (WMMA, M=64, K=512, N=256)
  k_cvt_bf16<<<cdiv_h((long long)N_GRAPH * 512, TPB), TPB, 0, stream>>>(pooled, PBF, N_GRAPH * 512);
  gemm_bf16_wmma<<<dim3(N_GRAPH / 16, H_DIM / 128), gblk, 0, stream>>>(
      PBF, Wc1T, z1, b_c1, N_GRAPH, 2 * H_DIM, H_DIM, GF_GELU);
  // z2 + log_softmax -> d_out
  k_cls_final<<<1, 64, 0, stream>>>(z1, W_c2, b_c2, out);
}
